// QuantizedLinear3bit_12257836662939
// MI455X (gfx1250) — compile-verified, matrix operand order checked by reading, not run on hardware
//
#include <hip/hip_runtime.h>
#include <hip/hip_bf16.h>

// ---------------------------------------------------------------------------
// QuantizedLinear3bit for MI455X (gfx1250, wave32, WMMA)
//   out[m,n] = (sum_k x[m,k] * qw[n,k]) * scale + bias[n]
//   M = B*S = 8192, K = 2048, N = 8192
// bf16 WMMA (weights exact in bf16), f32 accumulate, scale+bias f32 epilogue.
// Pipeline: register-staged global->LDS double pump; 32-bit saddr addressing.
// ---------------------------------------------------------------------------

typedef __bf16 v8bf  __attribute__((ext_vector_type(8)));
typedef __bf16 v16bf __attribute__((ext_vector_type(16)));
typedef float  v8f   __attribute__((ext_vector_type(8)));

#define M_TOT 8192
#define N_TOT 8192
#define K_TOT 2048

#define M_BLK 128
#define N_BLK 128
#define K_BLK 64
#define LDT   72   // LDS row stride in bf16 elems: 144 bytes, 16B aligned, bank-spread

// ------------------------- pack kernels ------------------------------------

__global__ __launch_bounds__(256) void q3_pack_x_kernel(
    const float* __restrict__ x, __bf16* __restrict__ xb) {
  size_t i = ((size_t)blockIdx.x * 256 + threadIdx.x) * 8;
  float4 f0 = *(const float4*)(x + i);
  float4 f1 = *(const float4*)(x + i + 4);
  v8bf r;
  r[0] = (__bf16)f0.x; r[1] = (__bf16)f0.y; r[2] = (__bf16)f0.z; r[3] = (__bf16)f0.w;
  r[4] = (__bf16)f1.x; r[5] = (__bf16)f1.y; r[6] = (__bf16)f1.z; r[7] = (__bf16)f1.w;
  *(v8bf*)(xb + i) = r;
}

__global__ __launch_bounds__(256) void q3_pack_w_kernel(
    const int* __restrict__ w, __bf16* __restrict__ wb) {
  size_t i = ((size_t)blockIdx.x * 256 + threadIdx.x) * 8;
  int4 q0 = *(const int4*)(w + i);
  int4 q1 = *(const int4*)(w + i + 4);
  v8bf r;  // ints in [-4,3] are exact in bf16
  r[0] = (__bf16)(float)q0.x; r[1] = (__bf16)(float)q0.y;
  r[2] = (__bf16)(float)q0.z; r[3] = (__bf16)(float)q0.w;
  r[4] = (__bf16)(float)q1.x; r[5] = (__bf16)(float)q1.y;
  r[6] = (__bf16)(float)q1.z; r[7] = (__bf16)(float)q1.w;
  *(v8bf*)(wb + i) = r;
}

// ------------------------- WMMA GEMM ---------------------------------------
// grid = (N/128, M/128), block = 256 threads = 8 waves (2 x 4 wave grid).
// Each wave computes a 64x32 patch = 4x2 tiles of 16x16.

__global__ __launch_bounds__(256) void q3_gemm_wmma_kernel(
    const __bf16* __restrict__ A,      // [M,K] bf16 activations
    const __bf16* __restrict__ Bw,     // [N,K] bf16 weights (K contiguous)
    const float*  __restrict__ scale_p,
    const float*  __restrict__ bias,
    float*        __restrict__ out) {
  __shared__ __bf16 lA[M_BLK * LDT];
  __shared__ __bf16 lB[N_BLK * LDT];

  const int t     = threadIdx.x;
  const int wave  = t >> 5;
  const int lane  = t & 31;
  const int waveM = wave >> 2;       // 0..1  -> 64-row strip
  const int waveN = wave & 3;        // 0..3  -> 32-col strip
  const int l16   = lane & 15;
  const int hh    = lane >> 4;       // half-wave select per ISA A/B layout

  const int bm = blockIdx.y * M_BLK;
  const int bn = blockIdx.x * N_BLK;

  // Block-uniform bases (SGPRs) so per-thread offsets stay 32-bit (saddr form).
  const __bf16* __restrict__ Ab = A  + (size_t)bm * K_TOT;
  const __bf16* __restrict__ Bb = Bw + (size_t)bn * K_TOT;

  const float scl = scale_p[0];

  // Per-thread staging offsets: 8-half chunks, 4 per thread per tile.
  unsigned goff[4], loff[4];
  #pragma unroll
  for (int i = 0; i < 4; ++i) {
    int idx = t + i * 256;              // 0..1023
    int row = idx >> 3;                 // 8 chunks per 64-half row
    int cik = (idx & 7) * 8;            // half offset within row
    goff[i] = (unsigned)(row * K_TOT + cik);
    loff[i] = (unsigned)(row * LDT + cik);
  }

  // Prologue: stage K-tile 0 into registers.
  v8bf sa[4], sb[4];
  #pragma unroll
  for (int i = 0; i < 4; ++i) {
    sa[i] = *(const v8bf*)(Ab + goff[i]);
    sb[i] = *(const v8bf*)(Bb + goff[i]);
  }

  v8f acc[4][2] = {};

  for (int kt = 0; kt < K_TOT; kt += K_BLK) {
    // Commit staged registers to LDS.
    #pragma unroll
    for (int i = 0; i < 4; ++i) {
      *(v8bf*)(&lA[loff[i]]) = sa[i];
      *(v8bf*)(&lB[loff[i]]) = sb[i];
    }
    __syncthreads();

    // Issue next tile's global loads now; they complete under the compute.
    if (kt + K_BLK < K_TOT) {
      const unsigned kn = (unsigned)(kt + K_BLK);
      #pragma unroll
      for (int i = 0; i < 4; ++i) {
        sa[i] = *(const v8bf*)(Ab + kn + goff[i]);
        sb[i] = *(const v8bf*)(Bb + kn + goff[i]);
      }
    }

    // L2-side prefetch two tiles ahead (global_prefetch_b8).
    if (kt + 2 * K_BLK < K_TOT) {
      int prow = t & 127;
      const __bf16* pp = ((t < 128) ? Ab : Bb)
                         + (size_t)prow * K_TOT + kt + 2 * K_BLK;
      __builtin_prefetch(pp, 0, 1);
    }

    // Two 16x16x32 K-steps per K-tile.
    #pragma unroll
    for (int ks = 0; ks < 2; ++ks) {
      const int ko = ks * 32 + 8 * hh;    // ISA 16-bit operand layout

      v16bf afrag[4];
      #pragma unroll
      for (int tm = 0; tm < 4; ++tm) {
        const __bf16* p = &lA[(waveM * 64 + tm * 16 + l16) * LDT + ko];
        v8bf lo = *(const v8bf*)p;        // K = base .. base+7
        v8bf hi = *(const v8bf*)(p + 16); // K = base+16 .. base+23
        afrag[tm] = __builtin_shufflevector(lo, hi,
            0, 1, 2, 3, 4, 5, 6, 7, 8, 9, 10, 11, 12, 13, 14, 15);
      }

      v16bf bfrag[2];
      #pragma unroll
      for (int tn = 0; tn < 2; ++tn) {
        const __bf16* p = &lB[(waveN * 32 + tn * 16 + l16) * LDT + ko];
        v8bf lo = *(const v8bf*)p;
        v8bf hi = *(const v8bf*)(p + 16);
        bfrag[tn] = __builtin_shufflevector(lo, hi,
            0, 1, 2, 3, 4, 5, 6, 7, 8, 9, 10, 11, 12, 13, 14, 15);
      }

      #pragma unroll
      for (int tm = 0; tm < 4; ++tm)
        #pragma unroll
        for (int tn = 0; tn < 2; ++tn)
          acc[tm][tn] = __builtin_amdgcn_wmma_f32_16x16x32_bf16(
              /*neg_a=*/false, afrag[tm], /*neg_b=*/false, bfrag[tn],
              /*c_mod=*/(short)0, acc[tm][tn],
              /*reuse_a=*/false, /*reuse_b=*/false);
    }

    __syncthreads();  // all waves done reading before next LDS commit
  }

  // Epilogue: D layout -> lane {0-15,16-31} = N, VGPR v = M rows {v, v+8}.
  float* __restrict__ outb = out + (size_t)bm * N_TOT + bn;
  const float* __restrict__ biasb = bias + bn;

  #pragma unroll
  for (int tm = 0; tm < 4; ++tm) {
    const int mloc = waveM * 64 + tm * 16 + 8 * hh;
    #pragma unroll
    for (int tn = 0; tn < 2; ++tn) {
      const int nloc = waveN * 32 + tn * 16 + l16;
      const float bv = biasb[nloc];
      #pragma unroll
      for (int v = 0; v < 8; ++v)
        outb[(unsigned)((mloc + v) * N_TOT + nloc)] = fmaf(acc[tm][tn][v], scl, bv);
    }
  }
}

// ------------------------- launcher ----------------------------------------

extern "C" void kernel_launch(void* const* d_in, const int* in_sizes, int n_in,
                              void* d_out, int out_size, void* d_ws, size_t ws_size,
                              hipStream_t stream) {
  (void)in_sizes; (void)n_in; (void)out_size; (void)ws_size;

  const float* x     = (const float*)d_in[0];   // [4,2048,2048] fp32
  const int*   qw    = (const int*)d_in[1];     // [8192,2048] int32 in [-4,3]
  const float* scale = (const float*)d_in[2];   // [1] fp32
  const float* bias  = (const float*)d_in[3];   // [8192] fp32
  float*       out   = (float*)d_out;           // [4,2048,8192] fp32

  __bf16* xb = (__bf16*)d_ws;                        // 32 MB
  __bf16* wb = xb + (size_t)M_TOT * K_TOT;           // 32 MB

  const size_t x_elems = (size_t)M_TOT * K_TOT;      // 16,777,216
  const size_t w_elems = (size_t)N_TOT * K_TOT;      // 16,777,216

  q3_pack_x_kernel<<<(int)(x_elems / (256 * 8)), 256, 0, stream>>>(x, xb);
  q3_pack_w_kernel<<<(int)(w_elems / (256 * 8)), 256, 0, stream>>>(qw, wb);

  dim3 grid(N_TOT / N_BLK, M_TOT / M_BLK);           // (64, 64)
  q3_gemm_wmma_kernel<<<grid, 256, 0, stream>>>(xb, wb, scale, bias, out);
}
